// GCN_adaboost_35871566856588
// MI455X (gfx1250) — matrix operand order repaired and checked by reading, not compiled, not run on hardware
//
#include <hip/hip_runtime.h>

// ---------------------------------------------------------------------------
// Types for WMMA
// ---------------------------------------------------------------------------
typedef __attribute__((ext_vector_type(16))) __bf16       v16bf;
typedef __attribute__((ext_vector_type(8)))  float        v8f;
typedef __attribute__((ext_vector_type(8)))  unsigned int v8u;

#define TILE_M 128
#define TILE_N 128
#define TILE_K 32
#define LDSK   34   // padded K stride (elements) -> 17-bank stride, conflict-light

// round-to-nearest-even f32 -> bf16 (ignores NaN corner case)
__device__ __forceinline__ unsigned short f2bf(float a) {
    union { float f; unsigned int u; } v; v.f = a;
    unsigned int x = v.u;
    return (unsigned short)((x + 0x7FFFu + ((x >> 16) & 1u)) >> 16);
}
__device__ __forceinline__ unsigned int pack2_bf16(float a, float b) {
    return (unsigned int)f2bf(a) | ((unsigned int)f2bf(b) << 16);
}

// ---------------------------------------------------------------------------
// C[M x N] = act( A[M x K] * B[K x N] + bias[N] )   (f32 in/out, bf16 WMMA)
// M % 128 == 0, N % 128 == 0, K % 32 == 0
// grid = (N/128, M/128), block = 256 threads (8 waves, wave32)
// Register double-buffered: global loads of tile k+1 overlap WMMA of tile k;
// L2 prefetch of tile k+2 (global_prefetch_b8).
// ---------------------------------------------------------------------------
__global__ __launch_bounds__(256) void wmma_gemm_bias_act(
    const float* __restrict__ A, const float* __restrict__ B,
    const float* __restrict__ bias, float* __restrict__ C,
    int M, int N, int K, int relu)
{
    __shared__ unsigned short lA[TILE_M * LDSK]; // [m][k], bf16 bits
    __shared__ unsigned short lB[TILE_N * LDSK]; // [n][k] (transposed), bf16 bits

    const int tid    = threadIdx.x;
    const int wave   = tid >> 5;
    const int lane   = tid & 31;
    const int half   = lane >> 4;   // 0: lanes 0-15, 1: lanes 16-31
    const int l16    = lane & 15;
    const int waveM  = (wave >> 2) * 64;  // 2 wave-rows
    const int waveN  = (wave & 3) * 32;   // 4 wave-cols
    const int blockM = blockIdx.y * TILE_M;
    const int blockN = blockIdx.x * TILE_N;

    // per-thread staging coordinates (fixed across k-steps)
    int arow[4], akq[4], bkr[4], bnq[4];
#pragma unroll
    for (int i = 0; i < 4; ++i) {
        int lin = tid + i * 256;     // 0..1023
        arow[i] = lin >> 3;          // 0..127 (A row)
        akq[i]  = lin & 7;           // float4 index along K
        bkr[i]  = lin >> 5;          // 0..31  (B row = K)
        bnq[i]  = lin & 31;          // float4 index along N
    }

    v8f acc[4][2];
    const v8f vzero = {0.f, 0.f, 0.f, 0.f, 0.f, 0.f, 0.f, 0.f};
#pragma unroll
    for (int i = 0; i < 4; ++i)
#pragma unroll
        for (int j = 0; j < 2; ++j) acc[i][j] = vzero;

    const int KT = K >> 5;

    float4 aReg[4], bReg[4];
    // prologue: fetch tile 0 into registers
#pragma unroll
    for (int i = 0; i < 4; ++i) {
        aReg[i] = ((const float4*)(A + (size_t)(blockM + arow[i]) * K))[akq[i]];
        bReg[i] = ((const float4*)(B + (size_t)bkr[i] * N + blockN))[bnq[i]];
    }

    for (int kt = 0; kt < KT; ++kt) {
        // ---- commit registers -> LDS (bf16)
#pragma unroll
        for (int i = 0; i < 4; ++i) {
            unsigned int p0 = pack2_bf16(aReg[i].x, aReg[i].y);
            unsigned int p1 = pack2_bf16(aReg[i].z, aReg[i].w);
            unsigned short* dst = &lA[arow[i] * LDSK + akq[i] * 4];
            *(unsigned int*)(dst)     = p0;
            *(unsigned int*)(dst + 2) = p1;

            lB[(bnq[i] * 4 + 0) * LDSK + bkr[i]] = f2bf(bReg[i].x);
            lB[(bnq[i] * 4 + 1) * LDSK + bkr[i]] = f2bf(bReg[i].y);
            lB[(bnq[i] * 4 + 2) * LDSK + bkr[i]] = f2bf(bReg[i].z);
            lB[(bnq[i] * 4 + 3) * LDSK + bkr[i]] = f2bf(bReg[i].w);
        }
        __syncthreads();

        // ---- issue global loads for tile kt+1 (overlap with WMMA below)
        if (kt + 1 < KT) {
            const int k1 = (kt + 1) * TILE_K;
#pragma unroll
            for (int i = 0; i < 4; ++i) {
                aReg[i] = ((const float4*)(A + (size_t)(blockM + arow[i]) * K + k1))[akq[i]];
                bReg[i] = ((const float4*)(B + (size_t)(k1 + bkr[i]) * N + blockN))[bnq[i]];
            }
        }
        // ---- L2 prefetch of tile kt+2 (global_prefetch_b8)
        if (kt + 2 < KT) {
            const int k2 = (kt + 2) * TILE_K;
            __builtin_prefetch(A + (size_t)(blockM + arow[0]) * K + k2 + akq[0] * 4, 0, 1);
            __builtin_prefetch(B + (size_t)(k2 + bkr[0]) * N + blockN + bnq[0] * 4, 0, 1);
        }

        // ---- A fragments: lane = M row, VGPR e holds K pair {2e(+8 if e>=4)} + 8*half
        v16bf afrag[4];
#pragma unroll
        for (int mi = 0; mi < 4; ++mi) {
            const int r = waveM + mi * 16 + l16;
            v8u au;
#pragma unroll
            for (int e = 0; e < 8; ++e) {
                int k = 2 * e + ((e >= 4) ? 8 : 0) + 8 * half;
                au[e] = *(const unsigned int*)&lA[r * LDSK + k];
            }
            afrag[mi] = __builtin_bit_cast(v16bf, au);
        }
        // ---- B fragments: lane = N col, VGPR e holds K pair {2e(+8 if e>=4)} + 8*half
        v16bf bfrag[2];
#pragma unroll
        for (int nj = 0; nj < 2; ++nj) {
            const int c = waveN + nj * 16 + l16;
            v8u bu;
#pragma unroll
            for (int e = 0; e < 8; ++e) {
                int k = 2 * e + ((e >= 4) ? 8 : 0) + 8 * half;
                bu[e] = *(const unsigned int*)&lB[c * LDSK + k];
            }
            bfrag[nj] = __builtin_bit_cast(v16bf, bu);
        }

#pragma unroll
        for (int mi = 0; mi < 4; ++mi)
#pragma unroll
            for (int nj = 0; nj < 2; ++nj)
                acc[mi][nj] = __builtin_amdgcn_wmma_f32_16x16x32_bf16(
                    false, afrag[mi], false, bfrag[nj],
                    (short)0, acc[mi][nj], false, false);

        __syncthreads();
    }

    // ---- epilogue: bias + optional relu, f32 store
#pragma unroll
    for (int nj = 0; nj < 2; ++nj) {
        const int bcol = blockN + waveN + nj * 16 + l16;
        const float bv = bias ? bias[bcol] : 0.0f;
#pragma unroll
        for (int mi = 0; mi < 4; ++mi) {
#pragma unroll
            for (int v = 0; v < 8; ++v) {
                const int row = blockM + waveM + mi * 16 + v + 8 * half;
                float val = acc[mi][nj][v] + bv;
                if (relu) val = fmaxf(val, 0.0f);
                C[(size_t)row * N + bcol] = val;
            }
        }
    }
}

// ---------------------------------------------------------------------------
// out[8192 x 16] = (maybe-relu(A) | concat A2) @ W[K x 16] + b[16]
// block = (16,16), grid = 8192/16
// ---------------------------------------------------------------------------
__global__ void dense16_kernel(const float* __restrict__ A,
                               const float* __restrict__ A2, int Kh,
                               const float* __restrict__ W,
                               const float* __restrict__ bias,
                               float* __restrict__ out, int relu_in)
{
    const int row = blockIdx.x * 16 + threadIdx.y;
    const int col = threadIdx.x;
    float s = bias[col];
    const float* a = A + (size_t)row * Kh;
    for (int k = 0; k < Kh; ++k) {
        float v = a[k];
        if (relu_in) v = fmaxf(v, 0.0f);
        s += v * W[k * 16 + col];
    }
    if (A2) {
        const float* a2 = A2 + (size_t)row * Kh;
        for (int k = 0; k < Kh; ++k)
            s += a2[k] * W[(Kh + k) * 16 + col];
    }
    out[(size_t)row * 16 + col] = s;
}

// ---------------------------------------------------------------------------
// adaboost alpha:  alpha = 0.5*log(s3/(sum-s3))
//   t3 = exp(-pred[index]*y[index]); gate = gatemat[index]*y[index]
//   s3 = sum(t3 where gate>=0); sum = sum(t3)
// single block of 256 threads
// ---------------------------------------------------------------------------
__global__ __launch_bounds__(256) void alpha_reduce_kernel(
    const float* __restrict__ pred, const float* __restrict__ gatemat,
    const int* __restrict__ index, const float* __restrict__ y,
    float* __restrict__ alpha_out)
{
    __shared__ float sA[256];
    __shared__ float sB[256];
    float s3 = 0.f, sall = 0.f;
    for (int e = threadIdx.x; e < 1024 * 16; e += 256) {
        int i = e >> 4, c = e & 15;
        int r = index[i];
        float yi = y[(size_t)r * 16 + c];
        float t3 = expf(-pred[(size_t)r * 16 + c] * yi);
        float t4 = gatemat[(size_t)r * 16 + c] * yi;
        sall += t3;
        if (t4 >= 0.f) s3 += t3;
    }
    sA[threadIdx.x] = s3;
    sB[threadIdx.x] = sall;
    __syncthreads();
    for (int off = 128; off > 0; off >>= 1) {
        if (threadIdx.x < off) {
            sA[threadIdx.x] += sA[threadIdx.x + off];
            sB[threadIdx.x] += sB[threadIdx.x + off];
        }
        __syncthreads();
    }
    if (threadIdx.x == 0)
        *alpha_out = 0.5f * logf(sA[0] / (sB[0] - sA[0]));
}

// out[i] = a[i] + b[i] * (*alpha)
__global__ void axpy_alpha_kernel(const float* __restrict__ a,
                                  const float* __restrict__ b,
                                  const float* __restrict__ alpha,
                                  float* __restrict__ out, int n)
{
    int i = blockIdx.x * 256 + threadIdx.x;
    if (i < n) out[i] = a[i] + b[i] * (*alpha);
}

// ---------------------------------------------------------------------------
// Host side
// ---------------------------------------------------------------------------
static void launch_gemm(const float* A, const float* B, const float* bias,
                        float* C, int M, int N, int K, int relu,
                        hipStream_t stream)
{
    dim3 grid((unsigned)(N / TILE_N), (unsigned)(M / TILE_M));
    wmma_gemm_bias_act<<<grid, 256, 0, stream>>>(A, B, bias, C, M, N, K, relu);
}

static void run_branch(const float* x, const float* adj,
                       const float* W1, const float* b1,
                       const float* W2, const float* b2,
                       const float* W3, const float* b3,
                       float* P, float* H, float* out, hipStream_t stream)
{
    const int N = 8192;
    launch_gemm(x,   W1, nullptr, P,   N, 512, 1024, 0, stream); // P  = x @ W1
    launch_gemm(adj, P,  b1,      H,   N, 512, N,    1, stream); // H1 = relu(adj@P + b1)
    launch_gemm(H,   W2, nullptr, P,   N, 512, 512,  0, stream); // P  = H1 @ W2
    launch_gemm(adj, P,  b2,      H,   N, 512, N,    1, stream); // H2 = relu(adj@P + b2)
    launch_gemm(H,   W3, nullptr, P,   N, 256, 512,  0, stream); // P  = H2 @ W3
    launch_gemm(adj, P,  b3,      out, N, 256, N,    0, stream); // out= adj@P + b3
}

extern "C" void kernel_launch(void* const* d_in, const int* in_sizes, int n_in,
                              void* d_out, int out_size, void* d_ws, size_t ws_size,
                              hipStream_t stream)
{
    const int N = 8192;
    const float* x    = (const float*)d_in[0];
    const float* adj1 = (const float*)d_in[1];
    const float* adj3 = (const float*)d_in[3];
    const float* adj4 = (const float*)d_in[4];
    const float* adj5 = (const float*)d_in[5];
    const float* y    = (const float*)d_in[6];
    const int*   idx  = (const int*)d_in[7];

    const float* gc1_W  = (const float*)d_in[8];  const float* gc1_b  = (const float*)d_in[9];
    const float* gc2_W  = (const float*)d_in[10]; const float* gc2_b  = (const float*)d_in[11];
    const float* gc3_W  = (const float*)d_in[12]; const float* gc3_b  = (const float*)d_in[13];
    const float* gc4_W  = (const float*)d_in[14]; const float* gc4_b  = (const float*)d_in[15];
    const float* gc5_W  = (const float*)d_in[16]; const float* gc5_b  = (const float*)d_in[17];
    const float* gc6_W  = (const float*)d_in[18]; const float* gc6_b  = (const float*)d_in[19];
    const float* gc10_W = (const float*)d_in[20]; const float* gc10_b = (const float*)d_in[21];
    const float* gc11_W = (const float*)d_in[22]; const float* gc11_b = (const float*)d_in[23];
    const float* gc12_W = (const float*)d_in[24]; const float* gc12_b = (const float*)d_in[25];
    const float* d1_W   = (const float*)d_in[26]; const float* d1_b   = (const float*)d_in[27];
    const float* d2_W   = (const float*)d_in[28]; const float* d2_b   = (const float*)d_in[29];
    const float* d3_W   = (const float*)d_in[30]; const float* d3_b   = (const float*)d_in[31];
    const float* d4_W   = (const float*)d_in[32]; const float* d4_b   = (const float*)d_in[33];
    const float* sim_W  = (const float*)d_in[34]; const float* sim_b  = (const float*)d_in[35];

    float* out = (float*)d_out;                  // [x2d | x3d | part2], 3 * 8192*16

    // workspace layout (floats)
    float* ws  = (float*)d_ws;
    float* P      = ws;                 // 8192*512
    float* H      = P  + (size_t)N * 512;
    float* x1     = H  + (size_t)N * 512;   // 8192*256 each
    float* x2     = x1 + (size_t)N * 256;
    float* x3     = x2 + (size_t)N * 256;
    float* x4     = x3 + (size_t)N * 256;
    float* x1d    = x4 + (size_t)N * 256;   // 8192*16 each
    float* x4d    = x1d + (size_t)N * 16;
    float* simd   = x4d + (size_t)N * 16;
    float* part1  = simd + (size_t)N * 16;
    float* alpha2 = part1 + (size_t)N * 16;
    float* alpha3 = alpha2 + 1;

    // 4 GCN branches (the heavy WMMA work)
    run_branch(x, adj5, gc1_W,  gc1_b,  gc2_W,  gc2_b,  gc3_W,  gc3_b,  P, H, x1, stream);
    run_branch(x, adj4, gc4_W,  gc4_b,  gc5_W,  gc5_b,  gc6_W,  gc6_b,  P, H, x2, stream);
    run_branch(x, adj3, gc4_W,  gc4_b,  gc5_W,  gc5_b,  gc6_W,  gc6_b,  P, H, x3, stream);
    run_branch(x, adj1, gc10_W, gc10_b, gc11_W, gc11_b, gc12_W, gc12_b, P, H, x4, stream);

    // dense heads (16-wide)
    dim3 dblk(16, 16);
    dense16_kernel<<<N / 16, dblk, 0, stream>>>(x1, nullptr, 256, d1_W, d1_b, x1d,          1);
    dense16_kernel<<<N / 16, dblk, 0, stream>>>(x2, nullptr, 256, d2_W, d2_b, out,          1);
    dense16_kernel<<<N / 16, dblk, 0, stream>>>(x3, nullptr, 256, d3_W, d3_b, out + N * 16, 1);
    dense16_kernel<<<N / 16, dblk, 0, stream>>>(x4, nullptr, 256, d4_W, d4_b, x4d,          1);
    dense16_kernel<<<N / 16, dblk, 0, stream>>>(x2, x3,      256, sim_W, sim_b, simd,       0);

    // adaboost stage 1: alpha2, part1 = x4d + simd*alpha2
    alpha_reduce_kernel<<<1, 256, 0, stream>>>(x4d, simd, idx, y, alpha2);
    axpy_alpha_kernel<<<(N * 16) / 256, 256, 0, stream>>>(x4d, simd, alpha2, part1, N * 16);

    // adaboost stage 2: alpha3, part2 = part1 + x1d*alpha3 -> third output
    alpha_reduce_kernel<<<1, 256, 0, stream>>>(part1, x1d, idx, y, alpha3);
    axpy_alpha_kernel<<<(N * 16) / 256, 256, 0, stream>>>(part1, x1d, alpha3,
                                                          out + 2 * N * 16, N * 16);
}
